// Inter_1_concat_attn_45578192945414
// MI455X (gfx1250) — matrix operand order repaired and checked by reading, not compiled, not run on hardware
//
#include <hip/hip_runtime.h>
#include <hip/hip_bf16.h>

// MI455X / gfx1250: wave32, WMMA bf16 16x16x32 (f32 accumulate).
// Pipeline: proj GEMM (WMMA, HBM-bound) -> attn MLP (WMMA) ->
// softmax/fuse -> classifier LN blocks (WMMA + shfl LayerNorm).
// Staging is split: global->VGPR loads issue BEFORE the WMMA chain,
// cvt+ds_store happen AFTER it, so HBM latency hides under matrix ops.
// B fragments are pipelined one tile ahead so s_wait_dscnt is non-zero.

typedef __bf16 bf16;
typedef __attribute__((ext_vector_type(16))) __bf16 bf16x16;
typedef __attribute__((ext_vector_type(8)))  __bf16 bf16x8;
typedef __attribute__((ext_vector_type(8)))  float  f32x8;

#define B_   16384
#define DM_  1024
#define H_   256
#define BKP  40   // 32-wide k chunk + 8 pad (80B row: 16B aligned, staggers banks)

__device__ __forceinline__ f32x8 wmma_bf16(bf16x16 a, bf16x16 b, f32x8 c) {
  return __builtin_amdgcn_wmma_f32_16x16x32_bf16(false, a, false, b, (short)0, c,
                                                 false, false);
}

// Build a 16-element bf16 fragment from two 16B LDS chunks (2x ds_load_b128).
__device__ __forceinline__ bf16x16 frag2(const bf16* p0, const bf16* p1) {
  bf16x8 lo = *(const bf16x8*)p0;
  bf16x8 hi = *(const bf16x8*)p1;
  bf16x16 r;
#pragma unroll
  for (int i = 0; i < 8; ++i) { r[i] = lo[i]; r[i + 8] = hi[i]; }
  return r;
}

// 16 consecutive f32 from global -> registers (4x global_load_b128).
__device__ __forceinline__ void gload16(const float* __restrict__ g, float* f) {
  *(float4*)(f + 0)  = ((const float4*)g)[0];
  *(float4*)(f + 4)  = ((const float4*)g)[1];
  *(float4*)(f + 8)  = ((const float4*)g)[2];
  *(float4*)(f + 12) = ((const float4*)g)[3];
}

// 16 f32 registers -> bf16 -> 32B to LDS (2x ds_store_b128).
__device__ __forceinline__ void cvt_store16(const float* f, bf16* __restrict__ l) {
  bf16x8 a, b;
#pragma unroll
  for (int i = 0; i < 8; ++i) { a[i] = (bf16)f[i]; b[i] = (bf16)f[i + 8]; }
  *(bf16x8*)(l)     = a;
  *(bf16x8*)(l + 8) = b;
}

// ---------------- K1: per-modality projection GEMM [B,1024] x [256,1024]^T ----
__global__ __launch_bounds__(256) void k_mri_proj(
    const float* __restrict__ t1, const float* __restrict__ t1c,
    const float* __restrict__ t2, const float* __restrict__ fl,
    const float* __restrict__ Wp, const float* __restrict__ bp,
    float* __restrict__ proj) {
  __shared__ __align__(16) bf16 As[2][128][BKP];
  __shared__ __align__(16) bf16 Bs[2][H_][BKP];
  const int m  = blockIdx.z;
  const int b0 = blockIdx.x * 128;
  const float* A = (m == 0) ? t1 : (m == 1) ? t1c : (m == 2) ? t2 : fl;
  const float* W = Wp + (size_t)m * H_ * DM_;
  const int t = threadIdx.x, wave = t >> 5, lane = t & 31;
  const int lidx = lane & 15, half = lane >> 4;
  const int arow = t >> 1, ako = (t & 1) * 16;  // this thread's A staging slot

  f32x8 acc[16];
#pragma unroll
  for (int i = 0; i < 16; ++i)
#pragma unroll
    for (int v = 0; v < 8; ++v) acc[i][v] = 0.f;

  float ra[16], rb[32];  // register staging (issued early, stored late)
  auto gloadA = [&](int k0) {
    gload16(A + (size_t)(b0 + arow) * DM_ + k0 + ako, ra);
  };
  auto gloadB = [&](int k0) {
    gload16(W + (size_t)t * DM_ + k0, rb);
    gload16(W + (size_t)t * DM_ + k0 + 16, rb + 16);
  };
  auto cstore = [&](int buf) {
    cvt_store16(ra, &As[buf][arow][ako]);
    cvt_store16(rb, &Bs[buf][t][0]);
    cvt_store16(rb + 16, &Bs[buf][t][16]);
  };

  const int row = wave * 16 + lidx;
  auto compute = [&](int cur) {
    bf16x16 afr = frag2(&As[cur][row][half * 16], &As[cur][row][half * 16 + 8]);
    const bf16* pb0 = &Bs[cur][lidx][half * 16];
    bf16x16 bcur = frag2(pb0, pb0 + 8);
#pragma unroll
    for (int nt = 0; nt < 16; ++nt) {
      bf16x16 bnext = bcur;
      if (nt + 1 < 16) {
        const bf16* pb = &Bs[cur][(nt + 1) * 16 + lidx][half * 16];
        bnext = frag2(pb, pb + 8);
      }
      acc[nt] = wmma_bf16(afr, bcur, acc[nt]);
      bcur = bnext;
    }
  };

  gloadA(0); gloadB(0);
  cstore(0);
  __syncthreads();

#pragma unroll 1
  for (int it = 0; it < DM_ / 32; ++it) {
    const int cur = it & 1;
    if (it + 1 < DM_ / 32) { gloadA((it + 1) * 32); gloadB((it + 1) * 32); }
    if (it + 2 < DM_ / 32)  // global_prefetch_b8 for the chunk after next
      __builtin_prefetch(A + (size_t)(b0 + arow) * DM_ + (it + 2) * 32, 0, 3);
    compute(cur);
    if (it + 1 < DM_ / 32) cstore(cur ^ 1);
    __syncthreads();
  }

#pragma unroll
  for (int nt = 0; nt < 16; ++nt) {
    const int n = nt * 16 + lidx;
    const float bias = bp[m * H_ + n];
#pragma unroll
    for (int v = 0; v < 8; ++v) {
      const int r = b0 + wave * 16 + v + 8 * half;   // C layout: row = v + 8*half
      proj[((size_t)r * 5 + m) * H_ + n] = acc[nt][v] + bias;
    }
  }
}

// ---------------- K2: clinical projection (K=8, VALU) ------------------------
__global__ __launch_bounds__(256) void k_clin(
    const float* __restrict__ fc, const float* __restrict__ W,
    const float* __restrict__ bias, float* __restrict__ proj) {
  const int i = blockIdx.x * 256 + threadIdx.x;
  const int b = i >> 8, h = i & 255;
  const float* x = fc + (size_t)b * 8;
  const float* w = W + (size_t)h * 8;
  float s = bias[h];
#pragma unroll
  for (int d = 0; d < 8; ++d) s += x[d] * w[d];
  proj[((size_t)b * 5 + 4) * H_ + h] = s;
}

// ---------------- K3: attention scores: Wa2 . tanh(proj @ Wa1^T + ba1) -------
__global__ __launch_bounds__(256) void k_attn(
    const float* __restrict__ proj, const float* __restrict__ Wa1,
    const float* __restrict__ ba1, const float* __restrict__ Wa2,
    const float* __restrict__ ba2, float* __restrict__ scores) {
  __shared__ __align__(16) bf16 As[128][BKP];
  __shared__ __align__(16) bf16 Bs[64][BKP];
  const int r0 = blockIdx.x * 128;
  const int t = threadIdx.x, wave = t >> 5, lane = t & 31;
  const int lidx = lane & 15, half = lane >> 4;
  const int row = wave * 16 + lidx;
  const int arow = t >> 1, ako = (t & 1) * 16;

  f32x8 acc[4];
#pragma unroll
  for (int i = 0; i < 4; ++i)
#pragma unroll
    for (int v = 0; v < 8; ++v) acc[i][v] = 0.f;

  float ra[16], rb[16];
  auto gload = [&](int k0) {
    gload16(proj + (size_t)(r0 + arow) * H_ + k0 + ako, ra);
    if (t < 128) gload16(Wa1 + (size_t)arow * H_ + k0 + ako, rb);
  };

  gload(0);
#pragma unroll 1
  for (int it = 0; it < H_ / 32; ++it) {
    __syncthreads();
    cvt_store16(ra, &As[arow][ako]);
    if (t < 128) cvt_store16(rb, &Bs[arow][ako]);
    __syncthreads();
    if (it + 1 < H_ / 32) gload((it + 1) * 32);
    bf16x16 afr = frag2(&As[row][half * 16], &As[row][half * 16 + 8]);
    const bf16* pb0 = &Bs[lidx][half * 16];
    bf16x16 bcur = frag2(pb0, pb0 + 8);
#pragma unroll
    for (int nt = 0; nt < 4; ++nt) {
      bf16x16 bnext = bcur;
      if (nt + 1 < 4) {
        const bf16* pb = &Bs[(nt + 1) * 16 + lidx][half * 16];
        bnext = frag2(pb, pb + 8);
      }
      acc[nt] = wmma_bf16(afr, bcur, acc[nt]);
      bcur = bnext;
    }
  }

  float part[8];
#pragma unroll
  for (int v = 0; v < 8; ++v) part[v] = 0.f;
#pragma unroll
  for (int nt = 0; nt < 4; ++nt) {
    const int n = nt * 16 + lidx;
    const float b1 = ba1[n], w2 = Wa2[n];
#pragma unroll
    for (int v = 0; v < 8; ++v) part[v] += tanhf(acc[nt][v] + b1) * w2;
  }
#pragma unroll
  for (int off = 1; off < 16; off <<= 1)
#pragma unroll
    for (int v = 0; v < 8; ++v) part[v] += __shfl_xor(part[v], off, 32);
  if (lidx == 0) {
    const float b2 = ba2[0];
#pragma unroll
    for (int v = 0; v < 8; ++v)
      scores[r0 + wave * 16 + v + 8 * half] = part[v] + b2;
  }
}

// ---------------- K4: softmax over 5 modalities + fused sum ------------------
__global__ __launch_bounds__(256) void k_fuse(
    const float* __restrict__ proj, const float* __restrict__ scores,
    float* __restrict__ fused, float* __restrict__ attn) {
  const int b = blockIdx.x, h = threadIdx.x;
  __shared__ float sraw[5];
  if (h < 5) sraw[h] = scores[b * 5 + h];
  __syncthreads();
  float s[5];
#pragma unroll
  for (int j = 0; j < 5; ++j) s[j] = sraw[j];
  float mx = s[0];
#pragma unroll
  for (int j = 1; j < 5; ++j) mx = fmaxf(mx, s[j]);
  float e[5], sum = 0.f;
#pragma unroll
  for (int j = 0; j < 5; ++j) { e[j] = expf(s[j] - mx); sum += e[j]; }
  const float inv = 1.f / sum;
  float a = 0.f;
#pragma unroll
  for (int j = 0; j < 5; ++j) a += (e[j] * inv) * proj[((size_t)b * 5 + j) * H_ + h];
  fused[(size_t)b * H_ + h] = a;
  if (h < 5) attn[b * 5 + h] = e[h] * inv;
}

// ---------------- K5: fused @ Wc1^T + bc1 -> LN(g1,be1) -> ReLU --------------
__global__ __launch_bounds__(256) void k_cls1(
    const float* __restrict__ X, const float* __restrict__ W,
    const float* __restrict__ bias, const float* __restrict__ g,
    const float* __restrict__ be, float* __restrict__ Y) {
  __shared__ __align__(16) bf16 As[128][BKP];
  __shared__ __align__(16) bf16 Bs[128][BKP];
  const int r0 = blockIdx.x * 128;
  const int t = threadIdx.x, wave = t >> 5, lane = t & 31;
  const int lidx = lane & 15, half = lane >> 4;
  const int row = wave * 16 + lidx;
  const int arow = t >> 1, ako = (t & 1) * 16;

  f32x8 acc[8];
#pragma unroll
  for (int i = 0; i < 8; ++i)
#pragma unroll
    for (int v = 0; v < 8; ++v) acc[i][v] = 0.f;

  float ra[16], rb[16];
  auto gload = [&](int k0) {
    gload16(X + (size_t)(r0 + arow) * H_ + k0 + ako, ra);
    gload16(W + (size_t)arow * H_ + k0 + ako, rb);
  };

  gload(0);
#pragma unroll 1
  for (int it = 0; it < H_ / 32; ++it) {
    __syncthreads();
    cvt_store16(ra, &As[arow][ako]);
    cvt_store16(rb, &Bs[arow][ako]);
    __syncthreads();
    if (it + 1 < H_ / 32) gload((it + 1) * 32);
    bf16x16 afr = frag2(&As[row][half * 16], &As[row][half * 16 + 8]);
    const bf16* pb0 = &Bs[lidx][half * 16];
    bf16x16 bcur = frag2(pb0, pb0 + 8);
#pragma unroll
    for (int nt = 0; nt < 8; ++nt) {
      bf16x16 bnext = bcur;
      if (nt + 1 < 8) {
        const bf16* pb = &Bs[(nt + 1) * 16 + lidx][half * 16];
        bnext = frag2(pb, pb + 8);
      }
      acc[nt] = wmma_bf16(afr, bcur, acc[nt]);
      bcur = bnext;
    }
  }

  float s[8], q[8];
#pragma unroll
  for (int v = 0; v < 8; ++v) { s[v] = 0.f; q[v] = 0.f; }
#pragma unroll
  for (int nt = 0; nt < 8; ++nt) {
    const float bb = bias[nt * 16 + lidx];
#pragma unroll
    for (int v = 0; v < 8; ++v) {
      const float x = acc[nt][v] + bb;
      acc[nt][v] = x; s[v] += x; q[v] += x * x;
    }
  }
#pragma unroll
  for (int off = 1; off < 16; off <<= 1)
#pragma unroll
    for (int v = 0; v < 8; ++v) {
      s[v] += __shfl_xor(s[v], off, 32);
      q[v] += __shfl_xor(q[v], off, 32);
    }
  float mean[8], rstd[8];
#pragma unroll
  for (int v = 0; v < 8; ++v) {
    mean[v] = s[v] * (1.f / 128.f);
    const float var = q[v] * (1.f / 128.f) - mean[v] * mean[v];
    rstd[v] = rsqrtf(var + 1e-5f);
  }
#pragma unroll
  for (int nt = 0; nt < 8; ++nt) {
    const int n = nt * 16 + lidx;
    const float gg = g[n], bbe = be[n];
#pragma unroll
    for (int v = 0; v < 8; ++v) {
      float y = (acc[nt][v] - mean[v]) * rstd[v] * gg + bbe;
      y = fmaxf(y, 0.f);
      Y[(size_t)(r0 + wave * 16 + v + 8 * half) * 128 + n] = y;
    }
  }
}

// ---------------- K6: h1 @ Wc2^T -> LN -> ReLU -> dot Wc3 + bc3 --------------
__global__ __launch_bounds__(256) void k_cls2(
    const float* __restrict__ X, const float* __restrict__ W,
    const float* __restrict__ bias, const float* __restrict__ g,
    const float* __restrict__ be, const float* __restrict__ Wc3,
    const float* __restrict__ bc3, float* __restrict__ out) {
  __shared__ __align__(16) bf16 As[128][BKP];
  __shared__ __align__(16) bf16 Bs[64][BKP];
  const int r0 = blockIdx.x * 128;
  const int t = threadIdx.x, wave = t >> 5, lane = t & 31;
  const int lidx = lane & 15, half = lane >> 4;
  const int row = wave * 16 + lidx;
  const int arow = t >> 1, ako = (t & 1) * 16;

  f32x8 acc[4];
#pragma unroll
  for (int i = 0; i < 4; ++i)
#pragma unroll
    for (int v = 0; v < 8; ++v) acc[i][v] = 0.f;

  float ra[16], rb[16];
  auto gload = [&](int k0) {
    gload16(X + (size_t)(r0 + arow) * 128 + k0 + ako, ra);
    if (t < 128) gload16(W + (size_t)arow * 128 + k0 + ako, rb);
  };

  gload(0);
#pragma unroll 1
  for (int it = 0; it < 128 / 32; ++it) {
    __syncthreads();
    cvt_store16(ra, &As[arow][ako]);
    if (t < 128) cvt_store16(rb, &Bs[arow][ako]);
    __syncthreads();
    if (it + 1 < 128 / 32) gload((it + 1) * 32);
    bf16x16 afr = frag2(&As[row][half * 16], &As[row][half * 16 + 8]);
    const bf16* pb0 = &Bs[lidx][half * 16];
    bf16x16 bcur = frag2(pb0, pb0 + 8);
#pragma unroll
    for (int nt = 0; nt < 4; ++nt) {
      bf16x16 bnext = bcur;
      if (nt + 1 < 4) {
        const bf16* pb = &Bs[(nt + 1) * 16 + lidx][half * 16];
        bnext = frag2(pb, pb + 8);
      }
      acc[nt] = wmma_bf16(afr, bcur, acc[nt]);
      bcur = bnext;
    }
  }

  float s[8], q[8];
#pragma unroll
  for (int v = 0; v < 8; ++v) { s[v] = 0.f; q[v] = 0.f; }
#pragma unroll
  for (int nt = 0; nt < 4; ++nt) {
    const float bb = bias[nt * 16 + lidx];
#pragma unroll
    for (int v = 0; v < 8; ++v) {
      const float x = acc[nt][v] + bb;
      acc[nt][v] = x; s[v] += x; q[v] += x * x;
    }
  }
#pragma unroll
  for (int off = 1; off < 16; off <<= 1)
#pragma unroll
    for (int v = 0; v < 8; ++v) {
      s[v] += __shfl_xor(s[v], off, 32);
      q[v] += __shfl_xor(q[v], off, 32);
    }
  float part[8];
#pragma unroll
  for (int v = 0; v < 8; ++v) {
    const float mean = s[v] * (1.f / 64.f);
    const float var  = q[v] * (1.f / 64.f) - mean * mean;
    const float rstd = rsqrtf(var + 1e-5f);
    s[v] = mean; q[v] = rstd; part[v] = 0.f;
  }
#pragma unroll
  for (int nt = 0; nt < 4; ++nt) {
    const int n = nt * 16 + lidx;
    const float gg = g[n], bbe = be[n], w3 = Wc3[n];
#pragma unroll
    for (int v = 0; v < 8; ++v) {
      float y = (acc[nt][v] - s[v]) * q[v] * gg + bbe;
      y = fmaxf(y, 0.f);
      part[v] += y * w3;
    }
  }
#pragma unroll
  for (int off = 1; off < 16; off <<= 1)
#pragma unroll
    for (int v = 0; v < 8; ++v) part[v] += __shfl_xor(part[v], off, 32);
  if (lidx == 0) {
    const float b3 = bc3[0];
#pragma unroll
    for (int v = 0; v < 8; ++v)
      out[r0 + wave * 16 + v + 8 * half] = part[v] + b3;
  }
}

extern "C" void kernel_launch(void* const* d_in, const int* in_sizes, int n_in,
                              void* d_out, int out_size, void* d_ws, size_t ws_size,
                              hipStream_t stream) {
  const float* t1     = (const float*)d_in[0];
  const float* t1c    = (const float*)d_in[1];
  const float* t2     = (const float*)d_in[2];
  const float* fl     = (const float*)d_in[3];
  const float* fc     = (const float*)d_in[4];
  const float* Wp_mri = (const float*)d_in[5];
  const float* bp_mri = (const float*)d_in[6];
  const float* Wp_cl  = (const float*)d_in[7];
  const float* bp_cl  = (const float*)d_in[8];
  const float* Wa1    = (const float*)d_in[9];
  const float* ba1    = (const float*)d_in[10];
  const float* Wa2    = (const float*)d_in[11];
  const float* ba2    = (const float*)d_in[12];
  const float* Wc1    = (const float*)d_in[13];
  const float* bc1    = (const float*)d_in[14];
  const float* g1     = (const float*)d_in[15];
  const float* be1    = (const float*)d_in[16];
  const float* Wc2    = (const float*)d_in[17];
  const float* bc2    = (const float*)d_in[18];
  const float* g2     = (const float*)d_in[19];
  const float* be2    = (const float*)d_in[20];
  const float* Wc3    = (const float*)d_in[21];
  const float* bc3    = (const float*)d_in[22];

  float* out  = (float*)d_out;     // [B] logits, then [B*5] attention weights
  float* attn = out + B_;

  float* proj   = (float*)d_ws;                       // [B,5,H]
  float* scores = proj + (size_t)B_ * 5 * H_;         // [B*5]
  float* fused  = scores + (size_t)B_ * 5;            // [B,H]
  float* h1     = fused + (size_t)B_ * H_;            // [B,128]

  k_mri_proj<<<dim3(B_ / 128, 1, 4), 256, 0, stream>>>(t1, t1c, t2, fl,
                                                       Wp_mri, bp_mri, proj);
  k_clin<<<(B_ * H_) / 256, 256, 0, stream>>>(fc, Wp_cl, bp_cl, proj);
  k_attn<<<(B_ * 5) / 128, 256, 0, stream>>>(proj, Wa1, ba1, Wa2, ba2, scores);
  k_fuse<<<B_, 256, 0, stream>>>(proj, scores, fused, attn);
  k_cls1<<<B_ / 128, 256, 0, stream>>>(fused, Wc1, bc1, g1, be1, h1);
  k_cls2<<<B_ / 128, 256, 0, stream>>>(h1, Wc2, bc2, g2, be2, Wc3, bc3, out);
}
